// SparseMoE_token_2018634629716
// MI455X (gfx1250) — matrix operand order repaired and checked by reading, not compiled, not run on hardware
//
#include <hip/hip_runtime.h>
#include <math.h>

typedef __bf16 bf16_t;
typedef __attribute__((ext_vector_type(16))) __bf16 v16bf;
typedef __attribute__((ext_vector_type(8)))  __bf16 v8bf;
typedef __attribute__((ext_vector_type(8)))  float  v8f;

#define D_DIM 768
#define H_DIM 3072
#define E_NUM 8
#define N_TOK 4096
#define MT    32      // tokens per workgroup
#define HC    128     // h-chunk width

// ---------------------------------------------------------------------------
// Fragment loaders (WMMA 16x16x32 bf16; A and B share this code's contiguous k order)
// A (16x32, 16-bit): lane m=lane&15; per lane 16 contiguous K values starting
//   at K = (lane>>4)*16  (two contiguous 16B chunks).
// B (32x16, 16-bit): lane n=lane&15; per lane 16 contiguous K values starting
//   at K = (lane>>4)*16  (two contiguous 16B chunks).
// C/D (16x16 f32): VGPR r, lane l -> M = r + 8*(l>>4), N = l&15.
// ---------------------------------------------------------------------------
__device__ __forceinline__ v16bf combine16(v8bf lo, v8bf hi) {
  return __builtin_shufflevector(lo, hi, 0,1,2,3,4,5,6,7,8,9,10,11,12,13,14,15);
}
__device__ __forceinline__ v16bf load_afrag(const bf16_t* rowk, int akb) {
  v8bf lo = *(const v8bf*)(rowk + akb);
  v8bf hi = *(const v8bf*)(rowk + 8 + akb);
  return combine16(lo, hi);
}
__device__ __forceinline__ v16bf load_bfrag(const bf16_t* p) {
  v8bf lo = *(const v8bf*)(p);
  v8bf hi = *(const v8bf*)(p + 8);
  return combine16(lo, hi);
}
__device__ __forceinline__ v8f wmma_bf16(v16bf a, v16bf b, v8f c) {
  return __builtin_amdgcn_wmma_f32_16x16x32_bf16(false, a, false, b, (short)0, c,
                                                 false, false);
}

// ---------------------------------------------------------------------------
// Router: one wave per token, fp32 exact path (top-k selection is discrete).
// ---------------------------------------------------------------------------
__global__ void __launch_bounds__(256) router_kernel(
    const float* __restrict__ x, const float* __restrict__ noise,
    const float* __restrict__ Wg, const float* __restrict__ bg,
    const float* __restrict__ Wn, const float* __restrict__ bn,
    float* __restrict__ probs) {
  const int wv   = threadIdx.x >> 5;
  const int lane = threadIdx.x & 31;
  const int t    = blockIdx.x * 8 + wv;
  const float* xr = x + (size_t)t * D_DIM;

  float ag[E_NUM], an[E_NUM];
#pragma unroll
  for (int e = 0; e < E_NUM; ++e) { ag[e] = 0.f; an[e] = 0.f; }

  for (int j = 0; j < D_DIM / 32; ++j) {
    int d = j * 32 + lane;
    float xv = xr[d];
#pragma unroll
    for (int e = 0; e < E_NUM; ++e) {
      ag[e] += xv * Wg[d * E_NUM + e];
      an[e] += xv * Wn[d * E_NUM + e];
    }
  }
#pragma unroll
  for (int off = 16; off >= 1; off >>= 1) {
#pragma unroll
    for (int e = 0; e < E_NUM; ++e) {
      ag[e] += __shfl_xor(ag[e], off, 32);
      an[e] += __shfl_xor(an[e], off, 32);
    }
  }
  if (lane == 0) {
    float nz[E_NUM];
#pragma unroll
    for (int e = 0; e < E_NUM; ++e) {
      float lg = ag[e] + bg[e];
      float sx = an[e] + bn[e];
      float sp = (sx > 20.f) ? sx : log1pf(expf(sx));
      nz[e] = lg + noise[(size_t)t * E_NUM + e] * sp;
    }
    int i1 = 0;
#pragma unroll
    for (int e = 1; e < E_NUM; ++e) if (nz[e] > nz[i1]) i1 = e;
    int i2 = (i1 == 0) ? 1 : 0;
#pragma unroll
    for (int e = 0; e < E_NUM; ++e) if (e != i1 && nz[e] > nz[i2]) i2 = e;
    float e2 = expf(nz[i2] - nz[i1]);
    float inv = 1.f / (1.f + e2);
#pragma unroll
    for (int e = 0; e < E_NUM; ++e) probs[(size_t)t * E_NUM + e] = 0.f;
    probs[(size_t)t * E_NUM + i1] = inv;
    probs[(size_t)t * E_NUM + i2] = e2 * inv;
  }
}

// ---------------------------------------------------------------------------
// Batched tiled transpose + fp32 -> bf16 convert: out[b][c*R + r] = in[b][r*C + c]
// ---------------------------------------------------------------------------
__global__ void transpose_to_bf16(const float* __restrict__ in,
                                  bf16_t* __restrict__ out, int R, int C) {
  __shared__ bf16_t tile[32][33];
  const size_t b = blockIdx.z;
  const float* ib = in + b * (size_t)R * C;
  bf16_t*      ob = out + b * (size_t)R * C;
  const int c0 = blockIdx.x * 32, r0 = blockIdx.y * 32;
  const int tx = threadIdx.x, ty = threadIdx.y;
#pragma unroll
  for (int k = 0; k < 4; ++k)
    tile[ty + 8 * k][tx] = (bf16_t)ib[(size_t)(r0 + ty + 8 * k) * C + c0 + tx];
  __syncthreads();
#pragma unroll
  for (int k = 0; k < 4; ++k)
    ob[(size_t)(c0 + ty + 8 * k) * R + r0 + tx] = tile[tx][ty + 8 * k];
}

// ---------------------------------------------------------------------------
// Fused experts: x-tile(32x768 bf16 in LDS) -> per (expert, h-chunk):
// GEMM1 (WMMA, K=768) + bias + exact GELU + gate-prob fold -> LDS h-tile ->
// GEMM2 partial (WMMA) into persistent 32x768 accumulators -> +gated b2 -> out
// ---------------------------------------------------------------------------
__global__ void __launch_bounds__(256, 1) moe_main(
    const float* __restrict__ x, const float* __restrict__ probs,
    const bf16_t* __restrict__ W1T,  // [E][H][D] bf16
    const bf16_t* __restrict__ W2T,  // [E][D][H] bf16
    const float* __restrict__ b1, const float* __restrict__ b2,
    float* __restrict__ out) {
  __shared__ bf16_t xs[MT * D_DIM];  // 48 KB
  __shared__ bf16_t hs[MT * HC];     // 8 KB
  __shared__ float  ps[MT * E_NUM];  // 1 KB

  const int tid   = threadIdx.x;
  const int wv    = tid >> 5;
  const int lane  = tid & 31;
  const int n     = lane & 15;
  const int hi16  = lane >> 4;
  const int akb   = hi16 * 16;  // A-frag per-lane K base
  const int bkb   = hi16 * 16;  // B-frag per-lane K base
  const int tbase = blockIdx.x * MT;

  for (int i = tid; i < MT * D_DIM; i += 256)
    xs[i] = (bf16_t)x[(size_t)tbase * D_DIM + i];
  if (tid < MT * E_NUM) ps[tid] = probs[(size_t)tbase * E_NUM + tid];
  __syncthreads();

  const v8f zv = {0.f, 0.f, 0.f, 0.f, 0.f, 0.f, 0.f, 0.f};
  v8f acc2[12];  // [nt 0..5][mt 0..1]: wave owns cols [wv*96, wv*96+96)
#pragma unroll
  for (int i = 0; i < 12; ++i) acc2[i] = zv;

  for (int e = 0; e < E_NUM; ++e) {
    const bf16_t* W1e = W1T + (size_t)e * H_DIM * D_DIM;
    const bf16_t* W2e = W2T + (size_t)e * D_DIM * H_DIM;
    for (int hc = 0; hc < H_DIM / HC; ++hc) {
      const int hbase = hc * HC;
      // ---- GEMM1: this wave computes 16 h-columns for the 32-token tile
      v8f acc1[2]; acc1[0] = zv; acc1[1] = zv;
      const int hcol = hbase + wv * 16 + n;
      const bf16_t* brow1 = W1e + (size_t)hcol * D_DIM + bkb;
#pragma unroll 4
      for (int ks = 0; ks < D_DIM / 32; ++ks) {
        v16bf bf = load_bfrag(brow1 + ks * 32);
        v16bf a0 = load_afrag(xs + (size_t)(n)      * D_DIM + ks * 32, akb);
        v16bf a1 = load_afrag(xs + (size_t)(16 + n) * D_DIM + ks * 32, akb);
        acc1[0] = wmma_bf16(a0, bf, acc1[0]);
        acc1[1] = wmma_bf16(a1, bf, acc1[1]);
      }
      // bias + exact GELU + fold gate prob, write bf16 h-tile to LDS
      const float b1v = b1[(size_t)e * H_DIM + hcol];
#pragma unroll
      for (int mt = 0; mt < 2; ++mt) {
#pragma unroll
        for (int r = 0; r < 8; ++r) {
          const int tl = mt * 16 + r + 8 * hi16;
          float zin = acc1[mt][r] + b1v;
          float g = 0.5f * zin * (1.0f + erff(zin * 0.70710678118f));
          hs[tl * HC + wv * 16 + n] = (bf16_t)(g * ps[tl * E_NUM + e]);
        }
      }
      __syncthreads();
      // ---- GEMM2 partial over this h-chunk
#pragma unroll
      for (int ks = 0; ks < HC / 32; ++ks) {
        v16bf a0 = load_afrag(hs + (size_t)(n)      * HC + ks * 32, akb);
        v16bf a1 = load_afrag(hs + (size_t)(16 + n) * HC + ks * 32, akb);
#pragma unroll
        for (int nt = 0; nt < 6; ++nt) {
          const int dcol = wv * 96 + nt * 16 + n;
          v16bf bf = load_bfrag(W2e + (size_t)dcol * H_DIM + hbase + ks * 32 + bkb);
          acc2[nt * 2 + 0] = wmma_bf16(a0, bf, acc2[nt * 2 + 0]);
          acc2[nt * 2 + 1] = wmma_bf16(a1, bf, acc2[nt * 2 + 1]);
        }
      }
      __syncthreads();
    }
  }
  // ---- epilogue: + sum_e p[t,e]*b2[e][d], store fp32
#pragma unroll
  for (int nt = 0; nt < 6; ++nt) {
    const int dcol = wv * 96 + nt * 16 + n;
    float b2v[E_NUM];
#pragma unroll
    for (int e = 0; e < E_NUM; ++e) b2v[e] = b2[(size_t)e * D_DIM + dcol];
#pragma unroll
    for (int mt = 0; mt < 2; ++mt) {
#pragma unroll
      for (int r = 0; r < 8; ++r) {
        const int tl = mt * 16 + r + 8 * hi16;
        float bias = 0.f;
#pragma unroll
        for (int e = 0; e < E_NUM; ++e) bias += ps[tl * E_NUM + e] * b2v[e];
        out[(size_t)(tbase + tl) * D_DIM + dcol] = acc2[nt * 2 + mt][r] + bias;
      }
    }
  }
}

// ---------------------------------------------------------------------------
extern "C" void kernel_launch(void* const* d_in, const int* in_sizes, int n_in,
                              void* d_out, int out_size, void* d_ws, size_t ws_size,
                              hipStream_t stream) {
  (void)in_sizes; (void)n_in; (void)out_size; (void)ws_size;
  const float* x     = (const float*)d_in[0];
  const float* noise = (const float*)d_in[1];
  const float* Wg    = (const float*)d_in[2];
  const float* bg    = (const float*)d_in[3];
  const float* Wn    = (const float*)d_in[4];
  const float* bn    = (const float*)d_in[5];
  const float* W1    = (const float*)d_in[6];
  const float* b1    = (const float*)d_in[7];
  const float* W2    = (const float*)d_in[8];
  const float* b2    = (const float*)d_in[9];
  float* out = (float*)d_out;

  char* ws = (char*)d_ws;
  float*  probs = (float*)ws;                                   // 128 KB
  bf16_t* W1T   = (bf16_t*)(ws + (size_t)131072);               // 36 MB
  bf16_t* W2T   = (bf16_t*)(ws + (size_t)131072 +
                            (size_t)E_NUM * D_DIM * H_DIM * 2); // 36 MB

  router_kernel<<<N_TOK / 8, 256, 0, stream>>>(x, noise, Wg, bg, Wn, bn, probs);
  transpose_to_bf16<<<dim3(H_DIM / 32, D_DIM / 32, E_NUM), dim3(32, 8), 0, stream>>>(
      W1, W1T, D_DIM, H_DIM);
  transpose_to_bf16<<<dim3(D_DIM / 32, H_DIM / 32, E_NUM), dim3(32, 8), 0, stream>>>(
      W2, W2T, H_DIM, D_DIM);
  moe_main<<<N_TOK / MT, 256, 0, stream>>>(x, probs, W1T, W2T, b1, b2, out);
}
